// S6_12824772346184
// MI455X (gfx1250) — matrix operand order repaired and checked by reading, not compiled, not run on hardware
//
#include <hip/hip_runtime.h>
#include <cmath>

typedef _Float16 half_t;
typedef __attribute__((ext_vector_type(16))) _Float16 v16h;
typedef __attribute__((ext_vector_type(8)))  _Float16 v8h;
typedef __attribute__((ext_vector_type(8)))  float    v8f;
typedef __attribute__((ext_vector_type(4)))  float    v4f;

#define DMODEL 64
#define ROWS   (8 * 2048)   // BATCH * SEQ_LEN = 16384
#define WPAD   72           // LDS-fallback padded K-stride (halves)
#define WQ_BYTES (3 * 64 * 64 * 2)   // 24 KB of f16 weights

__device__ __forceinline__ float softplus_f(float v) {
    // hardware trans ops (v_exp_f32 / v_log_f32); guard large args
    return (v > 20.f) ? v : __logf(1.f + __expf(v));
}

// ---------------------------------------------------------------------------
// Prep kernel: W1|W2|W3 (f32, row-major [n][k]) -> f16 at Wq[3][64][64].
// Row-major f16 means every WMMA B fragment is two runs of 8 contiguous halves, 16 apart.
// ---------------------------------------------------------------------------
__global__ void w_to_f16(const float* __restrict__ W1,
                         const float* __restrict__ W2,
                         const float* __restrict__ W3,
                         half_t* __restrict__ Wq)
{
    const int c = blockIdx.x * blockDim.x + threadIdx.x;   // 8 elements per thread
    if (c >= 3 * 64 * 64 / 8) return;
    const float* Wsrc[3] = {W1, W2, W3};
    const int p   = c >> 9;            // / 512
    const int off = (c & 511) * 8;
    const v4f f0 = *(const v4f*)(Wsrc[p] + off);
    const v4f f1 = *(const v4f*)(Wsrc[p] + off + 4);
    v8h h;
    #pragma unroll
    for (int j = 0; j < 4; ++j) { h[j] = (half_t)f0[j]; h[4 + j] = (half_t)f1[j]; }
    *(v8h*)(Wq + p * 4096 + off) = h;
}

// ---------------------------------------------------------------------------
// Main kernel, workspace-weight variant: one wave32 per workgroup,
// 1024 workgroups -> fills every SIMD. No LDS, no barriers.
// ---------------------------------------------------------------------------
__global__ __launch_bounds__(32)
void s6_fused_ws(const float* __restrict__ x,
                 const half_t* __restrict__ Wq,
                 const float* __restrict__ b1,
                 const float* __restrict__ b2,
                 const float* __restrict__ b3,
                 float* __restrict__ y)
{
    const int lane    = threadIdx.x;
    const int half_id = lane >> 4;
    const int ln16    = lane & 15;
    const int r0      = blockIdx.x * 16;

    // ---- A fragments: x rows r0..r0+15, K=64, WMMA 16-bit A layout ----
    // element i: k = t*32 + (i/8)*16 + half_id*8 + (i%8)
    const float* px = x + (size_t)(r0 + ln16) * DMODEL;
    v16h a[2];
    #pragma unroll
    for (int t = 0; t < 2; ++t) {
        #pragma unroll
        for (int g = 0; g < 2; ++g) {
            const int kb = t * 32 + g * 16 + half_id * 8;
            const v4f f0 = *(const v4f*)(px + kb);
            const v4f f1 = *(const v4f*)(px + kb + 4);
            #pragma unroll
            for (int j = 0; j < 4; ++j) {
                a[t][g * 8 + j]     = (half_t)f0[j];
                a[t][g * 8 + 4 + j] = (half_t)f1[j];
            }
        }
    }

    // B fragment: column n = nt*16+ln16, k = t*32+half_id*8+(0..7) and t*32+16+half_id*8+(0..7).
    // 24 KB weight set is WGP$-resident -> near-free re-reads.
    auto loadB = [&](int p, int nt, int t) -> v16h {
        const half_t* wp = Wq + ((p * 64 + nt * 16 + ln16) * 64 + t * 32 + half_id * 8);
        const v8h lo = *(const v8h*)wp;
        const v8h hi = *(const v8h*)(wp + 16);
        v16h b;
        #pragma unroll
        for (int j = 0; j < 8; ++j) { b[j] = lo[j]; b[8 + j] = hi[j]; }
        return b;
    };

    // ---- phase 1: B/C projections, accumulate elementwise product ----
    v8f pv = {};
    #pragma unroll
    for (int nt = 0; nt < 4; ++nt) {
        const float bb = b2[nt * 16 + ln16];
        const float bc = b3[nt * 16 + ln16];
        v8f cb, cc;
        #pragma unroll
        for (int r = 0; r < 8; ++r) { cb[r] = bb; cc[r] = bc; }
        cb = __builtin_amdgcn_wmma_f32_16x16x32_f16(false, a[0], false, loadB(1, nt, 0), (short)0, cb, false, false);
        cb = __builtin_amdgcn_wmma_f32_16x16x32_f16(false, a[1], false, loadB(1, nt, 1), (short)0, cb, false, false);
        cc = __builtin_amdgcn_wmma_f32_16x16x32_f16(false, a[0], false, loadB(2, nt, 0), (short)0, cc, false, false);
        cc = __builtin_amdgcn_wmma_f32_16x16x32_f16(false, a[1], false, loadB(2, nt, 1), (short)0, cc, false, false);
        pv += cb * cc;
    }

    // ---- phase 2: reduce over n within each 16-lane half -> s[m] ----
    #pragma unroll
    for (int r = 0; r < 8; ++r) {
        float v = pv[r];
        v += __shfl_xor(v, 1, 32);
        v += __shfl_xor(v, 2, 32);
        v += __shfl_xor(v, 4, 32);
        v += __shfl_xor(v, 8, 32);
        pv[r] = v;
    }

    // ---- phase 3: delta projection + softplus, y = x * delta * s ----
    const int m_lo = half_id * 8;
    #pragma unroll
    for (int nt = 0; nt < 4; ++nt) {
        const float bd = b1[nt * 16 + ln16];
        v8f cd;
        #pragma unroll
        for (int r = 0; r < 8; ++r) cd[r] = bd;
        cd = __builtin_amdgcn_wmma_f32_16x16x32_f16(false, a[0], false, loadB(0, nt, 0), (short)0, cd, false, false);
        cd = __builtin_amdgcn_wmma_f32_16x16x32_f16(false, a[1], false, loadB(0, nt, 1), (short)0, cd, false, false);
        const int d = nt * 16 + ln16;
        #pragma unroll
        for (int r = 0; r < 8; ++r) {
            const size_t off = (size_t)(r0 + m_lo + r) * DMODEL + d;
            y[off] = x[off] * softplus_f(cd[r]) * pv[r];
        }
    }
}

// ---------------------------------------------------------------------------
// Fallback (ws too small): LDS-staged weights, 4 waves / 256 blocks,
// vectorized f32->f16 preload (b128 loads, ds_store_b128).
// ---------------------------------------------------------------------------
__global__ __launch_bounds__(128)
void s6_fused_lds(const float* __restrict__ x,
                  const float* __restrict__ W1, const float* __restrict__ b1,
                  const float* __restrict__ W2, const float* __restrict__ b2,
                  const float* __restrict__ W3, const float* __restrict__ b3,
                  float* __restrict__ y)
{
    __shared__ __align__(16) half_t Wh[3][64 * WPAD];
    __shared__ float bsh[3][64];

    const int tid = threadIdx.x;
    const float* Wsrc[3] = {W1, W2, W3};
    // 1536 8-element chunks, 12 per thread
    #pragma unroll
    for (int it = 0; it < 12; ++it) {
        const int c   = tid + it * 128;
        const int p   = c >> 9;
        const int rem = c & 511;
        const int n   = rem >> 3;
        const int kb  = (rem & 7) * 8;
        const v4f f0 = *(const v4f*)(Wsrc[p] + n * 64 + kb);
        const v4f f1 = *(const v4f*)(Wsrc[p] + n * 64 + kb + 4);
        v8h h;
        #pragma unroll
        for (int j = 0; j < 4; ++j) { h[j] = (half_t)f0[j]; h[4 + j] = (half_t)f1[j]; }
        *(v8h*)(&Wh[p][n * WPAD + kb]) = h;
    }
    if (tid < 64) { bsh[0][tid] = b1[tid]; bsh[1][tid] = b2[tid]; bsh[2][tid] = b3[tid]; }
    __syncthreads();

    const int lane    = tid & 31;
    const int wave    = tid >> 5;
    const int half_id = lane >> 4;
    const int ln16    = lane & 15;
    const int r0      = (blockIdx.x * 4 + wave) * 16;

    const float* px = x + (size_t)(r0 + ln16) * DMODEL;
    v16h a[2];
    #pragma unroll
    for (int t = 0; t < 2; ++t) {
        #pragma unroll
        for (int g = 0; g < 2; ++g) {
            const int kb = t * 32 + g * 16 + half_id * 8;
            const v4f f0 = *(const v4f*)(px + kb);
            const v4f f1 = *(const v4f*)(px + kb + 4);
            #pragma unroll
            for (int j = 0; j < 4; ++j) {
                a[t][g * 8 + j]     = (half_t)f0[j];
                a[t][g * 8 + 4 + j] = (half_t)f1[j];
            }
        }
    }

    auto loadB = [&](int p, int nt, int t) -> v16h {
        const half_t* wp = &Wh[p][(nt * 16 + ln16) * WPAD + t * 32 + half_id * 8];
        const v8h lo = *(const v8h*)wp;
        const v8h hi = *(const v8h*)(wp + 16);
        v16h b;
        #pragma unroll
        for (int j = 0; j < 8; ++j) { b[j] = lo[j]; b[8 + j] = hi[j]; }
        return b;
    };

    v8f pv = {};
    #pragma unroll
    for (int nt = 0; nt < 4; ++nt) {
        const float bb = bsh[1][nt * 16 + ln16];
        const float bc = bsh[2][nt * 16 + ln16];
        v8f cb, cc;
        #pragma unroll
        for (int r = 0; r < 8; ++r) { cb[r] = bb; cc[r] = bc; }
        cb = __builtin_amdgcn_wmma_f32_16x16x32_f16(false, a[0], false, loadB(1, nt, 0), (short)0, cb, false, false);
        cb = __builtin_amdgcn_wmma_f32_16x16x32_f16(false, a[1], false, loadB(1, nt, 1), (short)0, cb, false, false);
        cc = __builtin_amdgcn_wmma_f32_16x16x32_f16(false, a[0], false, loadB(2, nt, 0), (short)0, cc, false, false);
        cc = __builtin_amdgcn_wmma_f32_16x16x32_f16(false, a[1], false, loadB(2, nt, 1), (short)0, cc, false, false);
        pv += cb * cc;
    }
    #pragma unroll
    for (int r = 0; r < 8; ++r) {
        float v = pv[r];
        v += __shfl_xor(v, 1, 32);
        v += __shfl_xor(v, 2, 32);
        v += __shfl_xor(v, 4, 32);
        v += __shfl_xor(v, 8, 32);
        pv[r] = v;
    }
    const int m_lo = half_id * 8;
    #pragma unroll
    for (int nt = 0; nt < 4; ++nt) {
        const float bd = bsh[0][nt * 16 + ln16];
        v8f cd;
        #pragma unroll
        for (int r = 0; r < 8; ++r) cd[r] = bd;
        cd = __builtin_amdgcn_wmma_f32_16x16x32_f16(false, a[0], false, loadB(0, nt, 0), (short)0, cd, false, false);
        cd = __builtin_amdgcn_wmma_f32_16x16x32_f16(false, a[1], false, loadB(0, nt, 1), (short)0, cd, false, false);
        const int d = nt * 16 + ln16;
        #pragma unroll
        for (int r = 0; r < 8; ++r) {
            const size_t off = (size_t)(r0 + m_lo + r) * DMODEL + d;
            y[off] = x[off] * softplus_f(cd[r]) * pv[r];
        }
    }
}

extern "C" void kernel_launch(void* const* d_in, const int* in_sizes, int n_in,
                              void* d_out, int out_size, void* d_ws, size_t ws_size,
                              hipStream_t stream) {
    const float* x  = (const float*)d_in[0];
    const float* W1 = (const float*)d_in[1];
    const float* b1 = (const float*)d_in[2];
    const float* W2 = (const float*)d_in[3];
    const float* b2 = (const float*)d_in[4];
    const float* W3 = (const float*)d_in[5];
    const float* b3 = (const float*)d_in[6];
    // d_in[7] (A) is mathematically dead: h starts at zero, so the dA*h term vanishes.
    float* y = (float*)d_out;

    if (ws_size >= WQ_BYTES) {
        half_t* Wq = (half_t*)d_ws;
        w_to_f16<<<6, 256, 0, stream>>>(W1, W2, W3, Wq);            // 24 KB f16 weights
        s6_fused_ws<<<ROWS / 16, 32, 0, stream>>>(x, Wq, b1, b2, b3, y);  // 1024 single-wave WGs
    } else {
        s6_fused_lds<<<ROWS / 64, 128, 0, stream>>>(x, W1, b1, W2, b2, W3, b3, y);
    }
}
